// Model_52467320488514
// MI455X (gfx1250) — compile-verified, matrix operand order checked by reading, not run on hardware
//
#include <hip/hip_runtime.h>
#include <math.h>

// ---------------------------------------------------------------------------
// Model dims
// ---------------------------------------------------------------------------
#define BB 128
#define TT 512
#define HH 256
#define CC 4
#define NWG_DIR 16     // workgroups per LSTM direction
#define HC_PER_WG 16   // hidden columns owned per workgroup
#define NLOC 64        // local gate columns per WG (4 gates * 16 hcols)

typedef __attribute__((ext_vector_type(16))) __bf16 v16bf;
typedef __attribute__((ext_vector_type(8)))  __bf16 v8bf;
typedef __attribute__((ext_vector_type(8)))  float  v8f;

// Load one 16-lane-shared half of a bf16 WMMA fragment: 16 contiguous bf16
// split as K[0..7] and K[16..23] relative to p (per the CDNA5 16-bit A/B
// VGPR layout; caller applies the per-lane koff).
static __device__ __forceinline__ v16bf load_frag16(const __bf16* p) {
    v8bf lo = *(const v8bf*)(p);
    v8bf hi = *(const v8bf*)(p + 16);
    return __builtin_shufflevector(lo, hi, 0, 1, 2, 3, 4, 5, 6, 7,
                                           8, 9, 10, 11, 12, 13, 14, 15);
}

static __device__ __forceinline__ float sigmoidf_(float x) {
    return 1.0f / (1.0f + __expf(-x));
}
// tanh via the fast hardware exp path (v_exp_f32)
static __device__ __forceinline__ float tanhf_(float x) {
    return 1.0f - 2.0f / (__expf(2.0f * x) + 1.0f);
}

// ---------------------------------------------------------------------------
// h0 = x * ff_w + ff_b, stored bf16 (B*T, H)
// ---------------------------------------------------------------------------
__global__ void ff_kernel(const float* __restrict__ x,
                          const float* __restrict__ ffw,
                          const float* __restrict__ ffb,
                          __bf16* __restrict__ X0) {
    size_t i = (size_t)blockIdx.x * 256 + threadIdx.x;
    if (i >= (size_t)BB * TT * HH) return;
    int h = (int)(i & (HH - 1));
    size_t bt = i >> 8;
    X0[i] = (__bf16)(x[bt] * ffw[h] + ffb[h]);
}

// ---------------------------------------------------------------------------
// Pack W = [Wih ; Whh] along K, bf16, rows = gate columns (4H, ldk).
// Also bias = bih + bhh.
// ---------------------------------------------------------------------------
__global__ void wconv_kernel(const float* __restrict__ wih,
                             const float* __restrict__ whh,
                             const float* __restrict__ bih,
                             const float* __restrict__ bhh,
                             int Kx, __bf16* __restrict__ W,
                             float* __restrict__ bias) {
    int ldk = Kx + HH;
    size_t total = (size_t)4 * HH * ldk;
    size_t i = (size_t)blockIdx.x * 256 + threadIdx.x;
    if (i >= total) return;
    int n = (int)(i / ldk);
    int k = (int)(i - (size_t)n * ldk);
    float v = (k < Kx) ? wih[(size_t)n * Kx + k]
                       : whh[(size_t)n * HH + (k - Kx)];
    W[i] = (__bf16)v;
    if (k == 0) bias[n] = bih[n] + bhh[n];
}

// ---------------------------------------------------------------------------
// Zero h ping-pong buffers (both directions) + both grid-sync counters
// ---------------------------------------------------------------------------
__global__ void lstm_init_kernel(__bf16* __restrict__ hbuf,
                                 unsigned* __restrict__ counter) {
    int i = blockIdx.x * 256 + threadIdx.x;
    if (i < 4 * BB * HH) hbuf[i] = (__bf16)0.0f;
    if (i == 0) { counter[0] = 0u; counter[32] = 0u; }
}

// ---------------------------------------------------------------------------
// Persistent bidirectional LSTM recurrence, both directions in one launch.
// grid = 2*NWG_DIR blocks x 256 threads; dir = blockIdx.x / NWG_DIR
// (dir 0 = forward weights/outputs, dir 1 = backward). Each WG owns 16 hidden
// columns (=> 64 gate columns). Per step:
//   A = [x_t | h_prev] (128 x ldk bf16) staged in LDS,
//   gates = A * W^T + bias via v_wmma_f32_16x16x32_bf16
//     (waves: 4 N-tiles x 2 M-halves; 4 accum fragments each),
//   c,h update with c resident in LDS, h exchanged via global bf16 ping-pong,
//   release/acquire fenced atomic grid barrier per direction.
// ---------------------------------------------------------------------------
__global__ void __launch_bounds__(256)
lstm_kernel(const __bf16* __restrict__ X, int Kx,
            const __bf16* __restrict__ Wall,     // 2 x (4H, ldk) bf16
            const float* __restrict__ biasall,   // 2 x 4H
            const int* __restrict__ lengths,
            float* __restrict__ outF,            // (B, T, H)
            float* __restrict__ outB,            // (B, T, H)
            __bf16* __restrict__ hball,          // 2 dirs x 2 ping x B*H
            unsigned* __restrict__ sync_counter) {
    const int ldk = Kx + HH;
    extern __shared__ char smem_raw[];
    __bf16* Alds  = (__bf16*)smem_raw;                               // 128*ldk
    float*  gates = (float*)(smem_raw + (size_t)BB * ldk * 2);       // 128*64
    float*  clds  = gates + BB * NLOC;                               // 128*16

    const int tid   = threadIdx.x;
    const int lane  = tid & 31;
    const int wv    = tid >> 5;        // wave 0..7
    const int l15   = lane & 15;
    const int lhalf = lane >> 4;       // 0 or 1
    const int koff  = lhalf ? 8 : 0;
    const int dir   = blockIdx.x / NWG_DIR;      // 0 = fwd, 1 = bwd
    const int wg    = blockIdx.x - dir * NWG_DIR;
    const int reverse = dir;

    const __bf16* W    = Wall + (size_t)dir * 4 * HH * ldk;
    const float*  bias = biasall + (size_t)dir * 4 * HH;
    float*        out  = dir ? outB : outF;
    __bf16*       hbuf = hball + (size_t)dir * 2 * BB * HH;
    unsigned* counter  = sync_counter + dir * 32;

    // wave tiling: nt = N-tile (16 gate cols), mh = M-half (64 batch rows)
    const int nt = wv & 3;
    const int mh = wv >> 2;
    const int jcol = nt * 16 + l15;                      // local col 0..63
    const int ncol = (jcol >> 4) * HH + wg * HC_PER_WG + (jcol & 15);
    const float bv = bias[ncol];
    const __bf16* Wrow = W + (size_t)ncol * ldk;

    // zero resident cell state
    for (int i = tid; i < BB * HC_PER_WG; i += 256) clds[i] = 0.0f;
    __syncthreads();

    const int chunks_per_row = ldk >> 3;              // 16B chunks per A row
    const int total_chunks   = BB * chunks_per_row;

    for (int s = 0; s < TT; ++s) {
        const __bf16* hprev = hbuf + (size_t)(s & 1) * BB * HH;
        __bf16*       hnext = hbuf + (size_t)((s & 1) ^ 1) * BB * HH;

        // ---- stage A = [x_t | h_prev] into LDS (per-row time gather) ----
        for (int ch = tid; ch < total_chunks; ch += 256) {
            int row = ch / chunks_per_row;
            int kc  = (ch - row * chunks_per_row) << 3;
            const uint4* src;
            if (kc < Kx) {
                int len = lengths[row];
                int te  = reverse ? ((s < len) ? (len - 1 - s) : s) : s;
                src = (const uint4*)(X + ((size_t)row * TT + te) * Kx + kc);
            } else {
                src = (const uint4*)(hprev + (size_t)row * HH + (kc - Kx));
            }
            ((uint4*)Alds)[ch] = *src;
        }
        __syncthreads();

        // ---- GEMM: 4 M-tiles (this wave's 64 rows) x lane's 16 cols ----
        v8f cf[4];
#pragma unroll
        for (int mt = 0; mt < 4; ++mt)
#pragma unroll
            for (int e = 0; e < 8; ++e) cf[mt][e] = bv;   // bias-preloaded C

        for (int kk = 0; kk < ldk; kk += 32) {
            // B fragment: column n of B == row n of W (contiguous)
            v16bf bfrag = load_frag16(Wrow + kk + koff);
            __builtin_prefetch(Wrow + kk + 32 + koff, 0, 1);
#pragma unroll
            for (int mt = 0; mt < 4; ++mt) {
                const __bf16* arow =
                    Alds + (size_t)(mh * 64 + mt * 16 + l15) * ldk + kk + koff;
                v16bf afrag = load_frag16(arow);
                cf[mt] = __builtin_amdgcn_wmma_f32_16x16x32_bf16(
                    false, afrag, false, bfrag, (short)0, cf[mt],
                    false, false);
            }
        }

        // ---- spill gates to LDS for the cross-gate c/h update ----
#pragma unroll
        for (int mt = 0; mt < 4; ++mt)
#pragma unroll
            for (int r = 0; r < 8; ++r)
                gates[(size_t)(mh * 64 + mt * 16 + r + lhalf * 8) * NLOC + jcol]
                    = cf[mt][r];
        __syncthreads();

        // ---- elementwise LSTM cell update for this WG's 16 hidden cols ----
        for (int it = tid; it < BB * HC_PER_WG; it += 256) {
            int b  = it >> 4;
            int hc = it & 15;
            float gi = gates[(size_t)b * NLOC + 0 * 16 + hc];
            float gf = gates[(size_t)b * NLOC + 1 * 16 + hc];
            float gg = gates[(size_t)b * NLOC + 2 * 16 + hc];
            float go = gates[(size_t)b * NLOC + 3 * 16 + hc];
            float c  = sigmoidf_(gf) * clds[it] + sigmoidf_(gi) * tanhf_(gg);
            clds[it] = c;
            float h  = sigmoidf_(go) * tanhf_(c);
            int len  = lengths[b];
            int te   = reverse ? ((s < len) ? (len - 1 - s) : s) : s;
            int hcg  = wg * HC_PER_WG + hc;
            out[((size_t)b * TT + te) * HH + hcg] = h;
            hnext[(size_t)b * HH + hcg] = (__bf16)h;
        }

        // ---- release/acquire fenced grid barrier (per direction) ----
        __builtin_amdgcn_fence(__ATOMIC_RELEASE, "agent");
        __syncthreads();
        if (tid == 0) {
            __hip_atomic_fetch_add(counter, 1u, __ATOMIC_RELEASE,
                                   __HIP_MEMORY_SCOPE_AGENT);
            unsigned target = (unsigned)(NWG_DIR * (s + 1));
            while (__hip_atomic_load(counter, __ATOMIC_ACQUIRE,
                                     __HIP_MEMORY_SCOPE_AGENT) < target) {
                __builtin_amdgcn_s_sleep(2);
            }
        }
        __syncthreads();
        __builtin_amdgcn_fence(__ATOMIC_ACQUIRE, "agent");
    }
}

// ---------------------------------------------------------------------------
// concat + mask: r = mask ? [out_f | out_b] : 0 ; also bf16 copy for next layer
// ---------------------------------------------------------------------------
__global__ void concat_kernel(const float* __restrict__ of,
                              const float* __restrict__ ob,
                              const int* __restrict__ lengths,
                              __bf16* __restrict__ Xn,
                              float* __restrict__ r) {
    size_t i = (size_t)blockIdx.x * 256 + threadIdx.x;
    size_t total = (size_t)BB * TT * 2 * HH;
    if (i >= total) return;
    int f = (int)(i & (2 * HH - 1));
    size_t bt = i >> 9;
    int b = (int)(bt >> 9);       // bt / TT, TT = 512
    int t = (int)(bt & (TT - 1));
    float v = 0.0f;
    if (t < lengths[b])
        v = (f < HH) ? of[bt * HH + f] : ob[bt * HH + (f - HH)];
    Xn[i] = (__bf16)v;
    r[i]  = v;
}

// ---------------------------------------------------------------------------
// pooling: mean / max / last over valid timesteps -> hpool (B, 6H)
// ---------------------------------------------------------------------------
__global__ void pool_kernel(const float* __restrict__ r,
                            const int* __restrict__ lengths,
                            float* __restrict__ hpool) {
    int i = blockIdx.x * 256 + threadIdx.x;
    if (i >= BB * 2 * HH) return;
    int b = i >> 9;
    int f = i & (2 * HH - 1);
    int len = lengths[b];
    const float* row = r + (size_t)b * TT * 2 * HH + f;
    float sum = 0.0f, mx = -INFINITY;
    for (int t = 0; t < len; ++t) {
        float v = row[(size_t)t * 2 * HH];
        sum += v;
        mx = fmaxf(mx, v);
    }
    float last = row[(size_t)(len - 1) * 2 * HH];
    hpool[(size_t)b * 1536 + f]        = sum / (float)len;
    hpool[(size_t)b * 1536 + 512 + f]  = mx;
    hpool[(size_t)b * 1536 + 1024 + f] = last;
}

// ---------------------------------------------------------------------------
// tiny dense + relu (conv branch)
// ---------------------------------------------------------------------------
__global__ void dense_relu_kernel(const float* __restrict__ in,
                                  const float* __restrict__ w,
                                  const float* __restrict__ bias,
                                  float* __restrict__ out, int K, int N) {
    int i = blockIdx.x * 256 + threadIdx.x;
    if (i >= BB * N) return;
    int b = i / N;
    int o = i - b * N;
    const float* xr = in + (size_t)b * K;
    const float* wr = w + (size_t)o * K;
    float acc = bias[o];
    for (int k = 0; k < K; ++k) acc += xr[k] * wr[k];
    out[i] = fmaxf(acc, 0.0f);
}

// ---------------------------------------------------------------------------
// final linear: out = [v(96) | hpool(1536)] @ lin_w^T + lin_b
// ---------------------------------------------------------------------------
__global__ void final_kernel(const float* __restrict__ v,
                             const float* __restrict__ hpool,
                             const float* __restrict__ lw,
                             const float* __restrict__ lb,
                             float* __restrict__ out) {
    int i = blockIdx.x * 256 + threadIdx.x;
    if (i >= BB * CC) return;
    int b = i / CC;
    int c = i - b * CC;
    const float* wr = lw + (size_t)c * (96 + 1536);
    float acc = lb[c];
    for (int k = 0; k < 96; ++k)   acc += v[(size_t)b * 96 + k] * wr[k];
    for (int k = 0; k < 1536; ++k) acc += hpool[(size_t)b * 1536 + k] * wr[96 + k];
    out[i] = acc;
}

// ---------------------------------------------------------------------------
// launcher
// ---------------------------------------------------------------------------
extern "C" void kernel_launch(void* const* d_in, const int* in_sizes, int n_in,
                              void* d_out, int out_size, void* d_ws, size_t ws_size,
                              hipStream_t stream) {
    const float* x       = (const float*)d_in[0];
    const int*   lengths = (const int*)d_in[1];
    const float* ff_w    = (const float*)d_in[2];
    const float* ff_b    = (const float*)d_in[3];
    const float* wih0f = (const float*)d_in[4];
    const float* whh0f = (const float*)d_in[5];
    const float* bih0f = (const float*)d_in[6];
    const float* bhh0f = (const float*)d_in[7];
    const float* wih0b = (const float*)d_in[8];
    const float* whh0b = (const float*)d_in[9];
    const float* bih0b = (const float*)d_in[10];
    const float* bhh0b = (const float*)d_in[11];
    const float* wih1f = (const float*)d_in[12];
    const float* whh1f = (const float*)d_in[13];
    const float* bih1f = (const float*)d_in[14];
    const float* bhh1f = (const float*)d_in[15];
    const float* wih1b = (const float*)d_in[16];
    const float* whh1b = (const float*)d_in[17];
    const float* bih1b = (const float*)d_in[18];
    const float* bhh1b = (const float*)d_in[19];
    const float* conv1_w = (const float*)d_in[20];
    const float* conv1_b = (const float*)d_in[21];
    const float* conv2_w = (const float*)d_in[22];
    const float* conv2_b = (const float*)d_in[23];
    const float* conv3_w = (const float*)d_in[24];
    const float* conv3_b = (const float*)d_in[25];
    const float* conv4_w = (const float*)d_in[26];
    const float* conv4_b = (const float*)d_in[27];
    const float* lin_w   = (const float*)d_in[28];
    const float* lin_b   = (const float*)d_in[29];
    float* outp = (float*)d_out;

    const size_t BT = (size_t)BB * TT;
    char* ws = (char*)d_ws;
    auto alloc = [&](size_t bytes) -> char* {
        char* p = ws;
        ws += (bytes + 255) & ~(size_t)255;
        return p;
    };
    __bf16*   X0    = (__bf16*)  alloc(BT * HH * 2);          // 32 MB
    __bf16*   X1    = (__bf16*)  alloc(BT * 2 * HH * 2);      // 64 MB
    float*    outF  = (float*)   alloc(BT * HH * 4);          // 64 MB
    float*    outB  = (float*)   alloc(BT * HH * 4);          // 64 MB
    float*    r     = (float*)   alloc(BT * 2 * HH * 4);      // 128 MB
    __bf16*   hbuf  = (__bf16*)  alloc((size_t)4 * BB * HH * 2);      // 2 dirs
    __bf16*   Wc    = (__bf16*)  alloc((size_t)2 * 4 * HH * 768 * 2); // 2 dirs
    float*    biasc = (float*)   alloc((size_t)2 * 4 * HH * 4);
    float*    hpool = (float*)   alloc((size_t)BB * 1536 * 4);
    float*    vA    = (float*)   alloc((size_t)BB * 512 * 4);
    float*    vB    = (float*)   alloc((size_t)BB * 512 * 4);
    unsigned* cnt   = (unsigned*)alloc(256);

    // allow >64KB dynamic LDS for the persistent LSTM kernel
    size_t smem_max = (size_t)BB * 768 * 2 + (size_t)BB * NLOC * 4
                    + (size_t)BB * HC_PER_WG * 4;
    (void)hipFuncSetAttribute((const void*)lstm_kernel,
                              hipFuncAttributeMaxDynamicSharedMemorySize,
                              (int)smem_max);

    // FF expansion -> bf16 layer-0 input
    {
        size_t total = BT * HH;
        ff_kernel<<<dim3((unsigned)((total + 255) / 256)), dim3(256), 0, stream>>>(
            x, ff_w, ff_b, X0);
    }

    auto run_bilstm = [&](const float* wihf, const float* whhf,
                          const float* bihf, const float* bhhf,
                          const float* wihb, const float* whhb,
                          const float* bihb, const float* bhhb,
                          int Kx, const __bf16* Xin) {
        int ldk = Kx + HH;
        size_t wtotal = (size_t)4 * HH * ldk;
        unsigned wblocks = (unsigned)((wtotal + 255) / 256);
        wconv_kernel<<<dim3(wblocks), dim3(256), 0, stream>>>(
            wihf, whhf, bihf, bhhf, Kx, Wc, biasc);
        wconv_kernel<<<dim3(wblocks), dim3(256), 0, stream>>>(
            wihb, whhb, bihb, bhhb, Kx,
            Wc + (size_t)4 * HH * ldk, biasc + 4 * HH);
        lstm_init_kernel<<<dim3(512), dim3(256), 0, stream>>>(hbuf, cnt);
        size_t smem = (size_t)BB * ldk * 2 + (size_t)BB * NLOC * 4
                    + (size_t)BB * HC_PER_WG * 4;
        lstm_kernel<<<dim3(2 * NWG_DIR), dim3(256), smem, stream>>>(
            Xin, Kx, Wc, biasc, lengths, outF, outB, hbuf, cnt);
    };

    // layer 0 (input K = H), both directions concurrently
    run_bilstm(wih0f, whh0f, bih0f, bhh0f,
               wih0b, whh0b, bih0b, bhh0b, HH, X0);
    {
        size_t total = BT * 2 * HH;
        concat_kernel<<<dim3((unsigned)((total + 255) / 256)), dim3(256), 0, stream>>>(
            outF, outB, lengths, X1, r);
    }

    // layer 1 (input K = 2H), both directions concurrently
    run_bilstm(wih1f, whh1f, bih1f, bhh1f,
               wih1b, whh1b, bih1b, bhh1b, 2 * HH, X1);
    {
        size_t total = BT * 2 * HH;
        concat_kernel<<<dim3((unsigned)((total + 255) / 256)), dim3(256), 0, stream>>>(
            outF, outB, lengths, X1, r);
    }

    // pooling
    pool_kernel<<<dim3((BB * 2 * HH) / 256), dim3(256), 0, stream>>>(r, lengths, hpool);

    // conv (dense) branch: 512 -> 64 -> 128 -> 256 -> 96
    dense_relu_kernel<<<dim3((BB * 64 + 255) / 256), dim3(256), 0, stream>>>(
        x, conv1_w, conv1_b, vA, 512, 64);
    dense_relu_kernel<<<dim3((BB * 128 + 255) / 256), dim3(256), 0, stream>>>(
        vA, conv2_w, conv2_b, vB, 64, 128);
    dense_relu_kernel<<<dim3((BB * 256 + 255) / 256), dim3(256), 0, stream>>>(
        vB, conv3_w, conv3_b, vA, 128, 256);
    dense_relu_kernel<<<dim3((BB * 96 + 255) / 256), dim3(256), 0, stream>>>(
        vA, conv4_w, conv4_b, vB, 256, 96);

    // final linear
    final_kernel<<<dim3((BB * CC + 255) / 256), dim3(256), 0, stream>>>(
        vB, hpool, lin_w, lin_b, outp);
}
